// JointPredictionLegalRGCN_33148557590801
// MI455X (gfx1250) — matrix operand order repaired and checked by reading, not compiled, not run on hardware
//
#include <hip/hip_runtime.h>

#define NN 100000
#define EE 1600000
#define FIN 770
#define KP1 800      // 770 padded to 25 k-steps of 32
#define HH 64

typedef __attribute__((ext_vector_type(16))) __bf16 v16bf;
typedef __attribute__((ext_vector_type(8)))  float  v8f;

union FragB {
  v16bf v;
  unsigned short u[16];
  uint4 q[2];
};

__device__ __forceinline__ unsigned short f2bf(float f) {
  union { float f; unsigned u; } c; c.f = f;
  unsigned u = c.u;
  return (unsigned short)((u + 0x7FFFu + ((u >> 16) & 1u)) >> 16);
}

__device__ __forceinline__ v8f wmma_bf16(v16bf a, v16bf b, v8f c) {
  // D = A(16x32 bf16) * B(32x16 bf16) + C(16x16 f32)
  return __builtin_amdgcn_wmma_f32_16x16x32_bf16(false, a, false, b, (short)0, c,
                                                 false, false);
}

__device__ __forceinline__ void loadB32(FragB& b, const unsigned short* p) {
  b.q[0] = *(const uint4*)p;
  b.q[1] = *(const uint4*)(p + 16);
}

// ---------------------------------------------------------------- utilities
__global__ __launch_bounds__(256) void zero_i32_kernel(int* p, int n) {
  int i = blockIdx.x * 256 + threadIdx.x;
  if (i < n) p[i] = 0;
}

__global__ __launch_bounds__(256) void count_edges_kernel(
    const int* __restrict__ dst, const int* __restrict__ et, int* __restrict__ cnt) {
  int e = blockIdx.x * 256 + threadIdx.x;
  if (e >= EE) return;
  atomicAdd(&cnt[et[e] * NN + dst[e]], 1);
}

__global__ __launch_bounds__(256) void inv_counts_kernel(
    const int* __restrict__ cnt, float* __restrict__ invc, int n) {
  int i = blockIdx.x * 256 + threadIdx.x;
  if (i >= n) return;
  int c = cnt[i];
  invc[i] = 1.0f / (float)(c > 0 ? c : 1);
}

__global__ __launch_bounds__(256) void sum_counts_kernel(
    const int* __restrict__ cnt, int* __restrict__ cntN) {
  int i = blockIdx.x * 256 + threadIdx.x;
  if (i >= NN) return;
  cntN[i] = cnt[i] + cnt[NN + i] + cnt[2 * NN + i];
}

// single-block exclusive scan over cntN[NN] -> offs[NN+1] (one-time CSR build)
__global__ __launch_bounds__(1024) void scan_kernel(
    const int* __restrict__ cntN, int* __restrict__ offs) {
  __shared__ int sdata[1024];
  __shared__ int carry;
  if (threadIdx.x == 0) carry = 0;
  __syncthreads();
  for (int base = 0; base < NN; base += 1024) {
    int i = base + threadIdx.x;
    int v = (i < NN) ? cntN[i] : 0;
    sdata[threadIdx.x] = v;
    __syncthreads();
    for (int off = 1; off < 1024; off <<= 1) {
      int t = (threadIdx.x >= off) ? sdata[threadIdx.x - off] : 0;
      __syncthreads();
      sdata[threadIdx.x] += t;
      __syncthreads();
    }
    int incl = sdata[threadIdx.x];
    if (i < NN) offs[i] = carry + incl - v;  // exclusive
    __syncthreads();
    if (threadIdx.x == 1023) carry += sdata[1023];
    __syncthreads();
  }
  if (threadIdx.x == 0) offs[NN] = carry;   // == EE
}

__global__ __launch_bounds__(256) void fill_csr_kernel(
    const int* __restrict__ dst, const int* __restrict__ offs,
    int* __restrict__ cursor, int* __restrict__ ebyDst) {
  int e = blockIdx.x * 256 + threadIdx.x;
  if (e >= EE) return;
  int d = dst[e];
  int w = atomicAdd(&cursor[d], 1);
  ebyDst[offs[d] + w] = e;
}

// Wcat^T [256][Kpad] bf16 from root [K,64] and W [3,K,64]; heads = root|W0|W1|W2
__global__ __launch_bounds__(256) void prep_wcat_kernel(
    const float* __restrict__ root, const float* __restrict__ W,
    unsigned short* __restrict__ out, int K, int Kpad) {
  int idx = blockIdx.x * 256 + threadIdx.x;
  if (idx >= 256 * Kpad) return;
  int j = idx / Kpad, k = idx % Kpad;
  float v = 0.f;
  if (k < K) {
    if (j < 64) v = root[k * 64 + j];
    else {
      int r = (j >> 6) - 1, jj = j & 63;
      v = W[((size_t)r * K + k) * 64 + jj];
    }
  }
  out[idx] = f2bf(v);
}

// transpose+pad: w [K,Cin] f32  ->  out [Cpad][K] bf16
__global__ __launch_bounds__(256) void prep_t_kernel(
    const float* __restrict__ w, unsigned short* __restrict__ out,
    int K, int Cin, int Cpad) {
  int idx = blockIdx.x * 256 + threadIdx.x;
  if (idx >= Cpad * K) return;
  int j = idx / K, k = idx % K;
  out[idx] = f2bf(j < Cin ? w[k * Cin + j] : 0.f);
}

// ------------------------------------------------- GEMM: layer-1 (f32 A, K=770)
// wave -> 16 rows x 128 cols (8 n-tiles); blockIdx.y selects col half.
__global__ __launch_bounds__(256) void gemm_l1_kernel(
    const float* __restrict__ X, const unsigned short* __restrict__ Bt,
    float* __restrict__ C) {
  const int lane = threadIdx.x & 31;
  const int wv   = threadIdx.x >> 5;
  const int mt   = blockIdx.x * 8 + wv;
  if (mt >= NN / 16) return;
  const int cg   = blockIdx.y;                 // 0,1 -> cols [0,128) / [128,256)
  const int m0 = mt * 16, lid = lane & 15, hlf = lane >> 4, khalf = hlf * 8;
  const float* rowA = X + (size_t)(m0 + lid) * FIN;
  const unsigned short* bcol = Bt + (size_t)(cg * 128 + lid) * KP1 + khalf;

  v8f zero = {};
  v8f acc[8];
#pragma unroll
  for (int i = 0; i < 8; ++i) acc[i] = zero;

  for (int ks = 0; ks < 25; ++ks) {
    const int k0 = ks * 32;
    FragB a;
    if (ks < 24) {
      const float* plo = rowA + k0 + khalf;
      const float* phi = plo + 16;
#pragma unroll
      for (int i = 0; i < 4; ++i) {
        float2 lo = *(const float2*)(plo + 2 * i);
        float2 hi = *(const float2*)(phi + 2 * i);
        a.u[2 * i + 0] = f2bf(lo.x);     a.u[2 * i + 1] = f2bf(lo.y);
        a.u[8 + 2 * i + 0] = f2bf(hi.x); a.u[8 + 2 * i + 1] = f2bf(hi.y);
      }
    } else {  // tail k-step: K 768..799, only 768/769 valid
#pragma unroll
      for (int i = 0; i < 8; ++i) {
        int ka = k0 + khalf + i, kb = ka + 16;
        a.u[i]     = (ka < FIN) ? f2bf(rowA[ka]) : (unsigned short)0;
        a.u[8 + i] = (kb < FIN) ? f2bf(rowA[kb]) : (unsigned short)0;
      }
    }
    FragB b[8];
#pragma unroll
    for (int nt = 0; nt < 8; ++nt)
      loadB32(b[nt], bcol + (size_t)nt * 16 * KP1 + k0);
#pragma unroll
    for (int nt = 0; nt < 8; ++nt)
      acc[nt] = wmma_bf16(a.v, b[nt].v, acc[nt]);
  }
#pragma unroll
  for (int nt = 0; nt < 8; ++nt)
#pragma unroll
    for (int v = 0; v < 8; ++v)
      C[(size_t)(m0 + v + 8 * hlf) * 256 + cg * 128 + nt * 16 + lid] = acc[nt][v];
}

// ------------------------------------------------- GEMM: layers 2/3 (bf16 A, K=64)
__global__ __launch_bounds__(256) void gemm_h64_kernel(
    const unsigned short* __restrict__ Xb, const unsigned short* __restrict__ Bt,
    float* __restrict__ C) {
  const int lane = threadIdx.x & 31;
  const int wv   = threadIdx.x >> 5;
  const int mt   = blockIdx.x * 8 + wv;
  if (mt >= NN / 16) return;
  const int cg   = blockIdx.y;
  const int m0 = mt * 16, lid = lane & 15, hlf = lane >> 4, khalf = hlf * 8;
  const unsigned short* rowA = Xb + (size_t)(m0 + lid) * 64;
  const unsigned short* bcol = Bt + (size_t)(cg * 128 + lid) * 64 + khalf;

  v8f zero = {};
  v8f acc[8];
#pragma unroll
  for (int i = 0; i < 8; ++i) acc[i] = zero;

#pragma unroll
  for (int ks = 0; ks < 2; ++ks) {
    const int k0 = ks * 32;
    FragB a;
    const unsigned short* pa = rowA + k0 + khalf;
    a.q[0] = *(const uint4*)pa;
    a.q[1] = *(const uint4*)(pa + 16);
    FragB b[8];
#pragma unroll
    for (int nt = 0; nt < 8; ++nt)
      loadB32(b[nt], bcol + (size_t)nt * 16 * 64 + k0);
#pragma unroll
    for (int nt = 0; nt < 8; ++nt)
      acc[nt] = wmma_bf16(a.v, b[nt].v, acc[nt]);
  }
#pragma unroll
  for (int nt = 0; nt < 8; ++nt)
#pragma unroll
    for (int v = 0; v < 8; ++v)
      C[(size_t)(m0 + v + 8 * hlf) * 256 + cg * 128 + nt * 16 + lid] = acc[nt][v];
}

// ---------------------------- fused CSR gather-aggregate + activation + bf16
// 64 threads per node (4 nodes/block). No atomics; per-node sum order fixed.
// mode 1: x1=relu (f32+bf16); mode 2: relu (bf16); mode 3: +resid (f32+bf16)
__global__ __launch_bounds__(256) void aggregate_kernel(
    const float* __restrict__ Ch, const float* __restrict__ bias,
    const int* __restrict__ offs, const int* __restrict__ ebyDst,
    const int* __restrict__ src, const int* __restrict__ et,
    const float* __restrict__ invc, const float* __restrict__ resid,
    float* __restrict__ xf_out, unsigned short* __restrict__ xb_out, int mode) {
  const int g = threadIdx.x >> 6, f = threadIdx.x & 63;
  const int n = blockIdx.x * 4 + g;
  if (n >= NN) return;
  float a = Ch[(size_t)n * 256 + f] + bias[f];     // root head + bias
  const float i0 = invc[n], i1 = invc[NN + n], i2 = invc[2 * NN + n];
  const int beg = offs[n], end = offs[n + 1];
  int e0 = 0, s0 = 0, r0 = 0;
  if (beg < end) { e0 = ebyDst[beg]; s0 = src[e0]; r0 = et[e0]; }
  for (int idx = beg; idx < end; ++idx) {
    int e1 = 0, s1 = 0, r1 = 0;
    if (idx + 1 < end) { e1 = ebyDst[idx + 1]; s1 = src[e1]; r1 = et[e1]; }
    float sc = (r0 == 0) ? i0 : ((r0 == 1) ? i1 : i2);
    a += Ch[(size_t)s0 * 256 + 64 + (r0 << 6) + f] * sc;
    e0 = e1; s0 = s1; r0 = r1;
  }
  int i = n * 64 + f;
  float v;
  if (mode < 3) v = a > 0.f ? a : 0.f;
  else          v = a + resid[i];
  if (xf_out) xf_out[i] = v;
  xb_out[i] = f2bf(v);
}

// ---------------------------------------------------------------- node MLP
__global__ __launch_bounds__(256) void node_mlp_kernel(
    const float* __restrict__ x3f, const float* __restrict__ nw1,
    const float* __restrict__ nb1, const float* __restrict__ nw2,
    const float* __restrict__ nb2, float* __restrict__ outN) {
  __shared__ float w1[64 * 32];
  __shared__ float w2[64];
  __shared__ float b1s[32];
  __shared__ float b2s[2];
  for (int i = threadIdx.x; i < 64 * 32; i += 256) w1[i] = nw1[i];
  for (int i = threadIdx.x; i < 64; i += 256) w2[i] = nw2[i];
  if (threadIdx.x < 32) b1s[threadIdx.x] = nb1[threadIdx.x];
  if (threadIdx.x < 2)  b2s[threadIdx.x] = nb2[threadIdx.x];
  __syncthreads();
  int n = blockIdx.x * 256 + threadIdx.x;
  if (n >= NN) return;
  float xv[64];
  const float4* xr = (const float4*)(x3f + (size_t)n * 64);
#pragma unroll
  for (int i = 0; i < 16; ++i) {
    float4 t = xr[i];
    xv[4 * i] = t.x; xv[4 * i + 1] = t.y; xv[4 * i + 2] = t.z; xv[4 * i + 3] = t.w;
  }
  float h[32];
#pragma unroll
  for (int j = 0; j < 32; ++j) h[j] = b1s[j];
  for (int f = 0; f < 64; ++f) {
    float xf = xv[f];
#pragma unroll
    for (int j = 0; j < 32; ++j) h[j] += xf * w1[f * 32 + j];
  }
  float o0 = b2s[0], o1 = b2s[1];
#pragma unroll
  for (int j = 0; j < 32; ++j) {
    float s = h[j] > 0.f ? h[j] : 0.f;
    o0 += s * w2[2 * j]; o1 += s * w2[2 * j + 1];
  }
  outN[2 * n] = o0; outN[2 * n + 1] = o1;
}

// --------------------------------------------- fused edge MLP (3 WMMA stages)
// wave -> 16 edges; concat(x3[src],x3[dst]) [16x128] @ ew1 -> relu -> @ew2 -> relu -> @ew3
__global__ __launch_bounds__(256) void edge_mlp_kernel(
    const unsigned short* __restrict__ x3b,
    const int* __restrict__ src, const int* __restrict__ dst,
    const unsigned short* __restrict__ E1T, const float* __restrict__ eb1,
    const unsigned short* __restrict__ E2T, const float* __restrict__ eb2,
    const unsigned short* __restrict__ E3T, const float* __restrict__ eb3,
    float* __restrict__ outE) {
  __shared__ __align__(16) unsigned short h1[8][16][64];
  __shared__ __align__(16) unsigned short h2[8][16][32];
  const int lane = threadIdx.x & 31;
  const int wv   = threadIdx.x >> 5;
  const int lid = lane & 15, hlf = lane >> 4, khalf = hlf * 8;
  const int e0 = (blockIdx.x * 8 + wv) * 16;          // E = 12500*8*16 exact
  const int si = src[e0 + lid];
  const int di = dst[e0 + lid];
  const unsigned short* rs = x3b + (size_t)si * 64;
  const unsigned short* rd = x3b + (size_t)di * 64;

  v8f zero = {};
  // stage 1: [16x128] @ ew1T -> 16x64
  v8f a1[4];
#pragma unroll
  for (int i = 0; i < 4; ++i) a1[i] = zero;
#pragma unroll
  for (int t = 0; t < 4; ++t) {
    const unsigned short* rowp = (t < 2) ? rs : rd;
    const unsigned short* pa = rowp + (t & 1) * 32 + khalf;
    FragB a;
    a.q[0] = *(const uint4*)pa;
    a.q[1] = *(const uint4*)(pa + 16);
    FragB b[4];
#pragma unroll
    for (int nt = 0; nt < 4; ++nt)
      loadB32(b[nt], E1T + (size_t)(nt * 16 + lid) * 128 + t * 32 + khalf);
#pragma unroll
    for (int nt = 0; nt < 4; ++nt)
      a1[nt] = wmma_bf16(a.v, b[nt].v, a1[nt]);
  }
#pragma unroll
  for (int nt = 0; nt < 4; ++nt) {
    float bias = eb1[nt * 16 + lid];
#pragma unroll
    for (int v = 0; v < 8; ++v) {
      float x = a1[nt][v] + bias;
      h1[wv][v + 8 * hlf][nt * 16 + lid] = f2bf(x > 0.f ? x : 0.f);
    }
  }
  __syncthreads();
  // stage 2: [16x64] @ ew2T -> 16x32
  v8f a2[2];
  a2[0] = zero; a2[1] = zero;
#pragma unroll
  for (int t = 0; t < 2; ++t) {
    const unsigned short* pa = &h1[wv][lid][t * 32 + khalf];
    FragB a;
    a.q[0] = *(const uint4*)pa;
    a.q[1] = *(const uint4*)(pa + 16);
    FragB b[2];
#pragma unroll
    for (int nt = 0; nt < 2; ++nt)
      loadB32(b[nt], E2T + (size_t)(nt * 16 + lid) * 64 + t * 32 + khalf);
#pragma unroll
    for (int nt = 0; nt < 2; ++nt)
      a2[nt] = wmma_bf16(a.v, b[nt].v, a2[nt]);
  }
#pragma unroll
  for (int nt = 0; nt < 2; ++nt) {
    float bias = eb2[nt * 16 + lid];
#pragma unroll
    for (int v = 0; v < 8; ++v) {
      float x = a2[nt][v] + bias;
      h2[wv][v + 8 * hlf][nt * 16 + lid] = f2bf(x > 0.f ? x : 0.f);
    }
  }
  __syncthreads();
  // stage 3: [16x32] @ ew3T(pad 3->16 cols) -> 16x16, cols 0..2 valid
  v8f a3 = zero;
  {
    const unsigned short* pa = &h2[wv][lid][khalf];
    FragB a;
    a.q[0] = *(const uint4*)pa;
    a.q[1] = *(const uint4*)(pa + 16);
    FragB b;
    loadB32(b, E3T + (size_t)lid * 32 + khalf);
    a3 = wmma_bf16(a.v, b.v, a3);
  }
  if (lid < 3) {
    float bias = eb3[lid];
#pragma unroll
    for (int v = 0; v < 8; ++v)
      outE[(size_t)(e0 + v + 8 * hlf) * 3 + lid] = a3[v] + bias;
  }
}

// ---------------------------------------------------------------- launcher
extern "C" void kernel_launch(void* const* d_in, const int* in_sizes, int n_in,
                              void* d_out, int out_size, void* d_ws, size_t ws_size,
                              hipStream_t stream) {
  const float* x     = (const float*)d_in[0];
  const int*   ei    = (const int*)d_in[1];
  const int*   etype = (const int*)d_in[2];
  const float* W1    = (const float*)d_in[3];
  const float* root1 = (const float*)d_in[4];
  const float* b1    = (const float*)d_in[5];
  const float* W2    = (const float*)d_in[6];
  const float* root2 = (const float*)d_in[7];
  const float* b2    = (const float*)d_in[8];
  const float* W3    = (const float*)d_in[9];
  const float* root3 = (const float*)d_in[10];
  const float* b3    = (const float*)d_in[11];
  const float* nw1   = (const float*)d_in[12];
  const float* nb1   = (const float*)d_in[13];
  const float* nw2   = (const float*)d_in[14];
  const float* nb2   = (const float*)d_in[15];
  const float* ew1   = (const float*)d_in[16];
  const float* eb1   = (const float*)d_in[17];
  const float* ew2   = (const float*)d_in[18];
  const float* eb2   = (const float*)d_in[19];
  const float* ew3   = (const float*)d_in[20];
  const float* eb3   = (const float*)d_in[21];
  const int* src = ei;
  const int* dst = ei + EE;

  // workspace layout (256B aligned blocks)
  size_t cur = 0;
  auto take = [&](size_t bytes) -> char* {
    char* p = (char*)d_ws + cur;
    cur += (bytes + 255) & ~(size_t)255;
    return p;
  };
  unsigned short* W1T   = (unsigned short*)take((size_t)256 * KP1 * 2);
  unsigned short* W2T   = (unsigned short*)take((size_t)256 * 64 * 2);
  unsigned short* W3T   = (unsigned short*)take((size_t)256 * 64 * 2);
  unsigned short* E1T   = (unsigned short*)take((size_t)64 * 128 * 2);
  unsigned short* E2T   = (unsigned short*)take((size_t)32 * 64 * 2);
  unsigned short* E3T   = (unsigned short*)take((size_t)16 * 32 * 2);
  int*            cnt   = (int*)take((size_t)3 * NN * 4);
  float*          invc  = (float*)take((size_t)3 * NN * 4);
  int*            cntN  = (int*)take((size_t)NN * 4);
  int*            offs  = (int*)take((size_t)(NN + 1) * 4);
  int*            curso = (int*)take((size_t)NN * 4);
  int*            ebyD  = (int*)take((size_t)EE * 4);
  float*          Ch    = (float*)take((size_t)NN * 256 * 4);
  float*          x1f   = (float*)take((size_t)NN * 64 * 4);
  float*          x3f   = (float*)take((size_t)NN * 64 * 4);
  unsigned short* x1b   = (unsigned short*)take((size_t)NN * 64 * 2);
  unsigned short* x2b   = (unsigned short*)take((size_t)NN * 64 * 2);
  unsigned short* x3b   = (unsigned short*)take((size_t)NN * 64 * 2);

  const dim3 B256(256);
  // ---- one-time graph prep: per-relation degree, inverse means, dst-CSR
  zero_i32_kernel<<<dim3((3 * NN + 255) / 256), B256, 0, stream>>>(cnt, 3 * NN);
  zero_i32_kernel<<<dim3((NN + 255) / 256), B256, 0, stream>>>(curso, NN);
  count_edges_kernel<<<dim3(EE / 256), B256, 0, stream>>>(dst, etype, cnt);
  inv_counts_kernel<<<dim3((3 * NN + 255) / 256), B256, 0, stream>>>(cnt, invc, 3 * NN);
  sum_counts_kernel<<<dim3((NN + 255) / 256), B256, 0, stream>>>(cnt, cntN);
  scan_kernel<<<dim3(1), dim3(1024), 0, stream>>>(cntN, offs);
  fill_csr_kernel<<<dim3(EE / 256), B256, 0, stream>>>(dst, offs, curso, ebyD);
  // ---- weights: transposed bf16
  prep_wcat_kernel<<<dim3(256 * KP1 / 256), B256, 0, stream>>>(root1, W1, W1T, FIN, KP1);
  prep_wcat_kernel<<<dim3(256 * 64 / 256), B256, 0, stream>>>(root2, W2, W2T, 64, 64);
  prep_wcat_kernel<<<dim3(256 * 64 / 256), B256, 0, stream>>>(root3, W3, W3T, 64, 64);
  prep_t_kernel<<<dim3(64 * 128 / 256), B256, 0, stream>>>(ew1, E1T, 128, 64, 64);
  prep_t_kernel<<<dim3(32 * 64 / 256), B256, 0, stream>>>(ew2, E2T, 64, 32, 32);
  prep_t_kernel<<<dim3(16 * 32 / 256), B256, 0, stream>>>(ew3, E3T, 32, 3, 16);

  const dim3 Ggemm((NN / 16 + 7) / 8, 2);    // 8 waves/block, 16 rows/wave, 2 col halves
  const dim3 Gagg(NN / 4);                   // 4 nodes/block, 64 threads/node

  // layer 1
  gemm_l1_kernel<<<Ggemm, B256, 0, stream>>>(x, W1T, Ch);
  aggregate_kernel<<<Gagg, B256, 0, stream>>>(Ch, b1, offs, ebyD, src, etype, invc,
                                              nullptr, x1f, x1b, 1);
  // layer 2
  gemm_h64_kernel<<<Ggemm, B256, 0, stream>>>(x1b, W2T, Ch);
  aggregate_kernel<<<Gagg, B256, 0, stream>>>(Ch, b2, offs, ebyD, src, etype, invc,
                                              nullptr, nullptr, x2b, 2);
  // layer 3 (+ residual x1)
  gemm_h64_kernel<<<Ggemm, B256, 0, stream>>>(x2b, W3T, Ch);
  aggregate_kernel<<<Gagg, B256, 0, stream>>>(Ch, b3, offs, ebyD, src, etype, invc,
                                              x1f, x3f, x3b, 3);

  // heads
  float* outN = (float*)d_out;
  float* outE = outN + (size_t)NN * 2;
  node_mlp_kernel<<<dim3((NN + 255) / 256), B256, 0, stream>>>(x3f, nw1, nb1, nw2, nb2, outN);
  edge_mlp_kernel<<<dim3(EE / 128), B256, 0, stream>>>(x3b, src, dst, E1T, eb1,
                                                       E2T, eb2, E3T, eb3, outE);
}
